// GATEncoder_85976655331725
// MI455X (gfx1250) — matrix operand order repaired and checked by reading, not compiled, not run on hardware
//
#include <hip/hip_runtime.h>
#include <hip/hip_bf16.h>

typedef __attribute__((ext_vector_type(16))) __bf16 v16bf;
typedef __attribute__((ext_vector_type(8)))  __bf16 v8bf;
typedef __attribute__((ext_vector_type(8)))  float  v8f;

// ---------- helpers ----------
__device__ __forceinline__ unsigned bf16bits(float f) {         // RNE, 16-bit result
  unsigned u = __float_as_uint(f);
  return (u + 0x7fffu + ((u >> 16) & 1u)) >> 16;
}
__device__ __forceinline__ __bf16 tobf(float f) {
  unsigned short s = (unsigned short)bf16bits(f);
  __bf16 b; __builtin_memcpy(&b, &s, 2); return b;
}
// order-preserving float <-> u32 encoding for atomic max
__device__ __forceinline__ unsigned fenc(float f) {
  unsigned u = __float_as_uint(f);
  return (u & 0x80000000u) ? ~u : (u | 0x80000000u);
}
__device__ __forceinline__ float fdec(unsigned e) {
  unsigned u = (e & 0x80000000u) ? (e & 0x7fffffffu) : ~e;
  return __uint_as_float(u);
}
__device__ __forceinline__ float lrelu(float v) { return v > 0.f ? v : 0.2f * v; }

// ---------- streaming f32 -> packed bf16 (2 per dword) ----------
__global__ void pack_bf16(const float* __restrict__ x, unsigned* __restrict__ y, int n2)
{
  int i = blockIdx.x * blockDim.x + threadIdx.x;
  if (i >= n2) return;
  float2 v = ((const float2*)x)[i];
  y[i] = (bf16bits(v.y) << 16) | bf16bits(v.x);
}

// ---------- GEMM: C[M x Ncols] = A[M x 128](bf16) * B[128 x Ncols](f32) ----------
__global__ __launch_bounds__(256) void gemm_wmma_bf16(
    const __bf16* __restrict__ A, const float* __restrict__ B,
    float* __restrict__ C, int M, int Ncols)
{
  const int lane = threadIdx.x & 31;
  const int wv   = threadIdx.x >> 5;     // 8 waves / block, one col-tile each
  const int half = lane >> 4;            // 0: lanes 0-15, 1: lanes 16-31
  const int l16  = lane & 15;
  const int col  = (blockIdx.y * 8 + wv) * 16 + l16;

  // Preload B fragments for all 4 K-steps (W is tiny; L2-resident).
  // B layout: lane L -> N = L%16, element i -> K = kstep*32 + half*16 + i.
  v16bf bfr[4];
  #pragma unroll
  for (int ks = 0; ks < 4; ++ks) {
    const int kbase = ks * 32 + half * 16;
    #pragma unroll
    for (int i = 0; i < 16; ++i)
      bfr[ks][i] = tobf(B[(size_t)(kbase + i) * Ncols + col]);
  }

  const int mtiles = M >> 4;             // M = 50000 -> 3125 tiles exactly
  for (int rt = blockIdx.x; rt < mtiles; rt += gridDim.x) {
    const __bf16* ap = A + (size_t)(rt * 16 + l16) * 128;  // A: lane row = M%16
    // two independent accumulators -> break D->C chains between adjacent WMMAs
    v8f acc0 = {}, acc1 = {};
    #pragma unroll
    for (int ks = 0; ks < 4; ++ks) {
      // A layout: elems 0-7 -> K = base..base+7 ; elems 8-15 -> K = base+8..base+15
      // base = ks*32 + half*16 ; each run of 8 bf16 = one 16B (b128) load
      const int base = ks * 32 + half * 16;
      v8bf lo = *(const v8bf*)(ap + base);
      v8bf hi = *(const v8bf*)(ap + base + 8);
      v16bf af = __builtin_shufflevector(lo, hi, 0, 1, 2, 3, 4, 5, 6, 7,
                                                 8, 9, 10, 11, 12, 13, 14, 15);
      if (ks & 1)
        acc1 = __builtin_amdgcn_wmma_f32_16x16x32_bf16(
            false, af, false, bfr[ks], (short)0, acc1, false, false);
      else
        acc0 = __builtin_amdgcn_wmma_f32_16x16x32_bf16(
            false, af, false, bfr[ks], (short)0, acc0, false, false);
    }
    // C/D layout: VGPR r -> M = rt*16 + half*8 + r, N = col
    float* cp = C + (size_t)(rt * 16 + half * 8) * Ncols + col;
    #pragma unroll
    for (int r = 0; r < 8; ++r) cp[(size_t)r * Ncols] = acc0[r] + acc1[r];
  }
}

// ---------- per-node attention logits: al[n,h] = sum_c h[n,h*C+c] * a[h*C+c] ----------
__global__ void node_logits(const float* __restrict__ h,
                            const float* __restrict__ a_src,
                            const float* __restrict__ a_dst,
                            float* __restrict__ als, float* __restrict__ ald,
                            int n, int C)
{
  int idx = blockIdx.x * blockDim.x + threadIdx.x;
  if (idx >= n * 4) return;
  int node = idx >> 2, hh = idx & 3;
  const float* hp = h + (size_t)node * 4 * C + hh * C;
  const float* as = a_src + hh * C;
  const float* ad = a_dst + hh * C;
  float s = 0.f, d = 0.f;
  for (int c = 0; c < C; ++c) { float v = hp[c]; s += v * as[c]; d += v * ad[c]; }
  als[idx] = s; ald[idx] = d;
}

// ---------- edge pass 1: segment max of leaky_relu logits ----------
__global__ void edge_max(const float* __restrict__ als, const float* __restrict__ ald,
                         const long long* __restrict__ ei, unsigned* __restrict__ menc,
                         int E, int EE)
{
  int idx = blockIdx.x * blockDim.x + threadIdx.x;
  if (idx >= EE * 4) return;
  int e = idx >> 2, hh = idx & 3;
  int src, dst;
  if (e < E) { src = (int)ei[e]; dst = (int)ei[E + e]; } else { src = dst = e - E; }
  float l = lrelu(als[src * 4 + hh] + ald[dst * 4 + hh]);
  atomicMax(&menc[dst * 4 + hh], fenc(l));
}

// ---------- edge pass 2: exp(logit - max), segment sum; cache exp per edge ----------
__global__ void edge_expsum(const float* __restrict__ als, const float* __restrict__ ald,
                            const long long* __restrict__ ei, const unsigned* __restrict__ menc,
                            float* __restrict__ ssum, float* __restrict__ expv,
                            int E, int EE)
{
  int idx = blockIdx.x * blockDim.x + threadIdx.x;
  if (idx >= EE * 4) return;
  int e = idx >> 2, hh = idx & 3;
  int src, dst;
  if (e < E) { src = (int)ei[e]; dst = (int)ei[E + e]; } else { src = dst = e - E; }
  float l = lrelu(als[src * 4 + hh] + ald[dst * 4 + hh]);
  float ev = __expf(l - fdec(menc[dst * 4 + hh]));
  expv[idx] = ev;
  atomicAdd(&ssum[dst * 4 + hh], ev);
}

// ---------- edge pass 3: out[dst] += alpha * h[src]  (wave per edge) ----------
__global__ void edge_aggregate(const float* __restrict__ h, const float* __restrict__ expv,
                               const float* __restrict__ ssum, const long long* __restrict__ ei,
                               float* __restrict__ out, int E, int EE, int C)
{
  int wave = (blockIdx.x * blockDim.x + threadIdx.x) >> 5;
  int lane = threadIdx.x & 31;
  if (wave >= EE) return;
  int src, dst;
  if (wave < E) { src = (int)ei[wave]; dst = (int)ei[E + wave]; } else { src = dst = wave - E; }
  const int F   = 4 * C;          // 128 or 256
  const int per = F >> 5;         // 4 or 8 floats per lane (one head per lane)
  const int f0  = lane * per;
  const int hh  = lane >> 3;      // = f0 / C for both C=32 and C=64
  const float coef = expv[(size_t)wave * 4 + hh] /
                     (ssum[(size_t)dst * 4 + hh] + 1e-16f);
  const float* hs = h + (size_t)src * F + f0;
  float* op = out + (size_t)dst * F + f0;
  for (int i = 0; i < per; i += 4) {
    float4 v = *(const float4*)(hs + i);
    atomicAdd(op + i + 0, v.x * coef);
    atomicAdd(op + i + 1, v.y * coef);
    atomicAdd(op + i + 2, v.z * coef);
    atomicAdd(op + i + 3, v.w * coef);
  }
}

// ---------- BatchNorm (training mode) ----------
__global__ void bn_stats(const float* __restrict__ x, float* __restrict__ stats,
                         int n, int F, int rows_per_blk)
{
  int c  = threadIdx.x;                       // blockDim.x == F
  int r0 = blockIdx.x * rows_per_blk;
  int r1 = min(n, r0 + rows_per_blk);
  float s = 0.f, s2 = 0.f;
  for (int r = r0; r < r1; ++r) { float v = x[(size_t)r * F + c]; s += v; s2 += v * v; }
  atomicAdd(&stats[c], s);
  atomicAdd(&stats[F + c], s2);
}

__global__ void bn_finalize(float* __restrict__ stats, const float* __restrict__ gamma,
                            const float* __restrict__ beta, int F, float invn)
{
  int c = threadIdx.x; if (c >= F) return;
  float mu  = stats[c] * invn;
  float var = stats[F + c] * invn - mu * mu;
  float sc  = gamma[c] * rsqrtf(var + 1e-5f);
  stats[2 * F + c] = sc;
  stats[3 * F + c] = beta[c] - mu * sc;
}

__global__ void bn_apply_relu(const float* __restrict__ x, const float* __restrict__ stats,
                              float* __restrict__ y, int n, int F)
{
  int idx = blockIdx.x * blockDim.x + threadIdx.x;
  if (idx >= n * F) return;
  int c = idx % F;
  float v = x[idx] * stats[2 * F + c] + stats[3 * F + c];
  y[idx] = v > 0.f ? v : 0.f;
}

// ---------- layer-2 epilogue: mean over heads + bias ----------
__global__ void head_mean_bias(const float* __restrict__ g, const float* __restrict__ bias,
                               float* __restrict__ y, int n)
{
  int idx = blockIdx.x * blockDim.x + threadIdx.x;
  if (idx >= n * 64) return;
  int node = idx >> 6, c = idx & 63;
  const float* p = g + (size_t)node * 256 + c;
  y[idx] = 0.25f * (p[0] + p[64] + p[128] + p[192]) + bias[c];
}

// ---------- orchestration ----------
extern "C" void kernel_launch(void* const* d_in, const int* in_sizes, int n_in,
                              void* d_out, int out_size, void* d_ws, size_t ws_size,
                              hipStream_t stream)
{
  const int N  = in_sizes[0] / 128;   // 50000
  const int E  = in_sizes[1] / 2;     // 800000
  const int EE = E + N;               // with self-loops

  const float*     x    = (const float*)d_in[0];
  const long long* ei   = (const long long*)d_in[1];
  const float*     W1   = (const float*)d_in[2];
  const float*     as1  = (const float*)d_in[3];
  const float*     ad1  = (const float*)d_in[4];
  // bias1 (d_in[5]) cancels under training-mode BN (per-column shift)
  const float*     g1   = (const float*)d_in[6];
  const float*     b1   = (const float*)d_in[7];
  const float*     W2   = (const float*)d_in[8];
  const float*     as2  = (const float*)d_in[9];
  const float*     ad2  = (const float*)d_in[10];
  const float*     bias2= (const float*)d_in[11];
  const float*     g2   = (const float*)d_in[12];
  const float*     b2   = (const float*)d_in[13];

  float* w = (float*)d_ws;
  // workspace layout (float offsets); Y aliases H1raw (dead by then)
  float*    H1raw = w;                                  // N*128
  float*    H1out = w + (size_t)N * 128;                // N*128 (agg -> BN in place)
  float*    H2raw = w + (size_t)N * 256;                // N*256
  float*    H2agg = w + (size_t)N * 512;                // N*256
  float*    expv  = w + (size_t)N * 768;                // EE*4
  float*    als   = expv + (size_t)EE * 4;              // N*4
  float*    ald   = als + (size_t)N * 4;                // N*4
  unsigned* menc  = (unsigned*)(ald + (size_t)N * 4);   // N*4
  float*    ssum  = (float*)(menc + (size_t)N * 4);     // N*4
  float*    stats = ssum + (size_t)N * 4;               // 4*128 (pad to 512)
  unsigned* Abf   = (unsigned*)(stats + 512);           // N*128 bf16 = N*64 dwords
  float*    Y     = H1raw;                              // N*64 alias

  const int THR = 256;
  const int gEdgeH = (EE * 4 + THR - 1) / THR;          // edge*head threads
  const int gEdgeW = (EE + 7) / 8;                      // wave per edge, 8 waves/blk
  const int gStats = (N + 255) / 256;
  const int gPack  = (N * 64 + THR - 1) / THR;          // N*128 floats / 2 per thread

  // ================= layer 1 =================
  pack_bf16<<<gPack, THR, 0, stream>>>(x, Abf, N * 64);
  gemm_wmma_bf16<<<dim3(512, 1), THR, 0, stream>>>((const __bf16*)Abf, W1, H1raw, N, 128);
  node_logits<<<(N * 4 + THR - 1) / THR, THR, 0, stream>>>(H1raw, as1, ad1, als, ald, N, 32);
  hipMemsetAsync(menc, 0, (size_t)N * 4 * 4, stream);   // enc(-NaN) floor
  hipMemsetAsync(ssum, 0, (size_t)N * 4 * 4, stream);
  hipMemsetAsync(H1out, 0, (size_t)N * 128 * 4, stream);
  edge_max<<<gEdgeH, THR, 0, stream>>>(als, ald, ei, menc, E, EE);
  edge_expsum<<<gEdgeH, THR, 0, stream>>>(als, ald, ei, menc, ssum, expv, E, EE);
  edge_aggregate<<<gEdgeW, THR, 0, stream>>>(H1raw, expv, ssum, ei, H1out, E, EE, 32);
  hipMemsetAsync(stats, 0, 4 * 128 * 4, stream);
  bn_stats<<<gStats, 128, 0, stream>>>(H1out, stats, N, 128, 256);
  bn_finalize<<<1, 128, 0, stream>>>(stats, g1, b1, 128, 1.f / (float)N);
  bn_apply_relu<<<(N * 128 + THR - 1) / THR, THR, 0, stream>>>(H1out, stats, H1out, N, 128);

  // ================= layer 2 =================
  pack_bf16<<<gPack, THR, 0, stream>>>(H1out, Abf, N * 64);
  gemm_wmma_bf16<<<dim3(512, 2), THR, 0, stream>>>((const __bf16*)Abf, W2, H2raw, N, 256);
  node_logits<<<(N * 4 + THR - 1) / THR, THR, 0, stream>>>(H2raw, as2, ad2, als, ald, N, 64);
  hipMemsetAsync(menc, 0, (size_t)N * 4 * 4, stream);
  hipMemsetAsync(ssum, 0, (size_t)N * 4 * 4, stream);
  hipMemsetAsync(H2agg, 0, (size_t)N * 256 * 4, stream);
  edge_max<<<gEdgeH, THR, 0, stream>>>(als, ald, ei, menc, E, EE);
  edge_expsum<<<gEdgeH, THR, 0, stream>>>(als, ald, ei, menc, ssum, expv, E, EE);
  edge_aggregate<<<gEdgeW, THR, 0, stream>>>(H2raw, expv, ssum, ei, H2agg, E, EE, 64);
  head_mean_bias<<<(N * 64 + THR - 1) / THR, THR, 0, stream>>>(H2agg, bias2, Y, N);
  hipMemsetAsync(stats, 0, 4 * 64 * 4, stream);
  bn_stats<<<gStats, 64, 0, stream>>>(Y, stats, N, 64, 256);
  bn_finalize<<<1, 64, 0, stream>>>(stats, g2, b2, 64, 1.f / (float)N);
  bn_apply_relu<<<(N * 64 + THR - 1) / THR, THR, 0, stream>>>(Y, stats, (float*)d_out, N, 64);
}
